// Attention_4346506904245
// MI455X (gfx1250) — matrix operand order repaired and checked by reading, not compiled, not run on hardware
//
#include <hip/hip_runtime.h>
#include <hip/hip_bf16.h>

typedef __attribute__((ext_vector_type(16))) __bf16 bf16x16;
typedef __attribute__((ext_vector_type(8)))  float  f32x8;

#define HEADS  16
#define DHEAD  64
#define SEQ    2048
#define DMODEL 1024
#define INNER  1024
#define BATCH  4

// ------------------------------------------------ async global->LDS support
#if defined(__has_builtin)
#  if __has_builtin(__builtin_amdgcn_global_load_async_to_lds_b128)
#    define HAVE_ASYNC_LDS 1
#  endif
#  if __has_builtin(__builtin_amdgcn_s_wait_asynccnt)
#    define HAVE_WAIT_ASYNC 1
#  endif
#endif
#ifndef HAVE_ASYNC_LDS
#  define HAVE_ASYNC_LDS 0
#endif
#ifndef HAVE_WAIT_ASYNC
#  define HAVE_WAIT_ASYNC 0
#endif

#if HAVE_ASYNC_LDS
typedef int v4i_ __attribute__((vector_size(16)));
typedef __attribute__((address_space(1))) v4i_ gv4i;   // global (HIP __device__) AS
typedef __attribute__((address_space(3))) v4i_ lv4i;   // LDS AS
#endif

// copy 16 bytes global -> LDS (async when available)
__device__ __forceinline__ void copy16_g2l(const unsigned short* g, unsigned short* l) {
#if HAVE_ASYNC_LDS
    __builtin_amdgcn_global_load_async_to_lds_b128((gv4i*)g, (lv4i*)l, 0, 0);
#else
    *(uint4*)l = *(const uint4*)g;
#endif
}

__device__ __forceinline__ void wait_async_copies() {
#if HAVE_ASYNC_LDS
#  if HAVE_WAIT_ASYNC
    __builtin_amdgcn_s_wait_asynccnt(0);
#  else
    asm volatile("s_wait_asynccnt 0x0" ::: "memory");
#  endif
#endif
}

// ---------------------------------------------------------------- helpers
__device__ __forceinline__ unsigned short f32_to_bf16(float f) {
    unsigned int u = __builtin_bit_cast(unsigned int, f);
    unsigned int r = u + 0x7FFFu + ((u >> 16) & 1u);   // round-to-nearest-even
    return (unsigned short)(r >> 16);
}

// A-fragment (16x32 bf16): lane = row (lane&15); k-pairs j=0..7 -> 2*hi8+2j,
// hi8=(lane>>4)*8 (contiguous k chunk of 16). p = start of this lane's row.
__device__ __forceinline__ bf16x16 load_afrag(const unsigned short* p, int hi8) {
    union { bf16x16 v; unsigned int u[8]; } f;
#pragma unroll
    for (int j = 0; j < 8; ++j) {
        int k0 = (hi8 << 1) + (j << 1);
        f.u[j] = *(const unsigned int*)(p + k0);
    }
    return f.v;
}

// B-fragment (32x16 bf16): lane = column (lane&15), k chunk = (lane>>4)*16.
// p = row of a k-major [n][k] tile -> 8 contiguous dwords.
__device__ __forceinline__ bf16x16 load_bfrag(const unsigned short* p, int hi16) {
    union { bf16x16 v; unsigned int u[8]; } f;
#pragma unroll
    for (int j = 0; j < 8; ++j)
        f.u[j] = *(const unsigned int*)(p + hi16 + 2 * j);
    return f.v;
}

__device__ __forceinline__ f32x8 wmma_bf16(bf16x16 a, bf16x16 b, f32x8 c) {
    return __builtin_amdgcn_wmma_f32_16x16x32_bf16(false, a, false, b,
                                                   (short)0, c, false, false);
}

// ------------------------------------------------ one-time conversion kernels
__global__ __launch_bounds__(256)
void cvt_bf16_kernel(const float* __restrict__ src, unsigned short* __restrict__ dst) {
    size_t base = ((size_t)blockIdx.x * 256 + threadIdx.x) * 8;
    float4 a = *(const float4*)(src + base);
    float4 b = *(const float4*)(src + base + 4);
    union { unsigned short h[8]; uint4 u; } o;
    o.h[0] = f32_to_bf16(a.x); o.h[1] = f32_to_bf16(a.y);
    o.h[2] = f32_to_bf16(a.z); o.h[3] = f32_to_bf16(a.w);
    o.h[4] = f32_to_bf16(b.x); o.h[5] = f32_to_bf16(b.y);
    o.h[6] = f32_to_bf16(b.z); o.h[7] = f32_to_bf16(b.w);
    *(uint4*)(dst + base) = o.u;
}

// W [K][N] f32 -> WT [N][K] bf16 (32x32 LDS tile transpose)
__global__ __launch_bounds__(256)
void transpose_cvt_kernel(const float* __restrict__ W, unsigned short* __restrict__ WT,
                          int Kdim, int Ndim) {
    __shared__ float tile[32][33];
    const int bx = blockIdx.x * 32;       // N base
    const int by = blockIdx.y * 32;       // K base
    const int x  = threadIdx.x & 31;
    const int y8 = threadIdx.x >> 5;      // 0..7
#pragma unroll
    for (int i = 0; i < 4; ++i) {
        int r = y8 + i * 8;
        tile[r][x] = W[(size_t)(by + r) * Ndim + bx + x];
    }
    __syncthreads();
#pragma unroll
    for (int i = 0; i < 4; ++i) {
        int r = y8 + i * 8;
        WT[(size_t)(bx + r) * Kdim + by + x] = f32_to_bf16(tile[x][r]);
    }
}

// ---------------------------------------------------------------- GEMM
// C(MxN) = A(MxK, bf16 row-major) @ W(KxN given as WT[N][K] bf16).
// mode 0: f32 row-major -> outF
// mode 1: bf16*scale   -> outQ in [B,h,S,d]
// mode 2: bf16*scale   -> outQ in [B,h,d,S]  (transposed, for V)
#define GBM 128
#define GBN 64
#define GBK 32
#define ASTR 40   // halves; 80B rows: 16B-aligned & conflict-free (40r mod 64 distinct)
#define TSTR 40

__global__ __launch_bounds__(256)
void gemm_bf16(const unsigned short* __restrict__ A, const unsigned short* __restrict__ WTg,
               float* __restrict__ outF, unsigned short* __restrict__ outQ,
               int M, int N, int K, int mode, float scale)
{
    __shared__ unsigned short As[2][GBM * ASTR];  // [row][k]
    __shared__ unsigned short Ws[2][GBN * TSTR];  // [col][k]

    const int tid  = threadIdx.x;
    const int lane = tid & 31;
    const int wv   = tid >> 5;
    const int hi   = lane >> 4;
    const int ln   = lane & 15;
    const int rowBase = blockIdx.y * GBM;
    const int colBase = blockIdx.x * GBN;

    // staging geometry (one 16B chunk per descriptor)
    const int arow0 = tid >> 2,        akoff = (tid & 3) * 8;   // As: chunks 0..255
    const int arow1 = (tid + 256) >> 2;                          //     chunks 256..511
    const int wcol  = tid >> 2,        wkoff = (tid & 3) * 8;   // Ws: 256 chunks

    f32x8 acc[4] = {};

    // prologue: stage k-block 0 into buffer 0
    copy16_g2l(A + (size_t)(rowBase + arow0) * K + akoff, &As[0][arow0 * ASTR + akoff]);
    copy16_g2l(A + (size_t)(rowBase + arow1) * K + akoff, &As[0][arow1 * ASTR + akoff]);
    copy16_g2l(WTg + (size_t)(colBase + wcol) * K + wkoff, &Ws[0][wcol * TSTR + wkoff]);
    wait_async_copies();
    __syncthreads();

    for (int kb = 0; kb < K; kb += GBK) {
        const int buf = (kb >> 5) & 1;
        if (kb + GBK < K) {          // prefetch next tile into other buffer
            const int nkb = kb + GBK;
            copy16_g2l(A + (size_t)(rowBase + arow0) * K + nkb + akoff,
                       &As[buf ^ 1][arow0 * ASTR + akoff]);
            copy16_g2l(A + (size_t)(rowBase + arow1) * K + nkb + akoff,
                       &As[buf ^ 1][arow1 * ASTR + akoff]);
            copy16_g2l(WTg + (size_t)(colBase + wcol) * K + nkb + wkoff,
                       &Ws[buf ^ 1][wcol * TSTR + wkoff]);
        }

        const unsigned short* ap = &As[buf][(wv * 16 + ln) * ASTR];
        bf16x16 af = load_afrag(ap, hi * 8);
#pragma unroll
        for (int t = 0; t < 4; ++t) {
            bf16x16 bf = load_bfrag(&Ws[buf][(t * 16 + ln) * TSTR], hi * 16);
            acc[t] = wmma_bf16(af, bf, acc[t]);
        }

        wait_async_copies();
        __syncthreads();
    }

    union { f32x8 v; float s[8]; } cu;
#pragma unroll
    for (int t = 0; t < 4; ++t) {
        cu.v = acc[t];
        int col = colBase + t * 16 + ln;
#pragma unroll
        for (int r = 0; r < 8; ++r) {
            int grow = rowBase + wv * 16 + r + hi * 8;
            float val = cu.s[r] * scale;
            if (mode == 0) {
                outF[(size_t)grow * N + col] = val;
            } else {
                int b = grow >> 11, s = grow & (SEQ - 1);
                int h = col >> 6,  dd = col & (DHEAD - 1);
                size_t idx = (mode == 1)
                    ? ((size_t)(b * HEADS + h) * SEQ + s) * DHEAD + dd     // [B,h,S,d]
                    : ((size_t)(b * HEADS + h) * DHEAD + dd) * SEQ + s;    // [B,h,d,S]
                outQ[idx] = f32_to_bf16(val);
            }
        }
    }
}

// ---------------------------------------------------------------- attention
// Flash-style; Q/K in [B,h,S,d] bf16, V pre-transposed [B,h,d,S] bf16.
#define ATT_BR 128
#define ATT_BC 32
#define KSTR   72    // [32 keys][64+8 d]  : 144B rows, 16B aligned, conflict-free
#define VTSTR  40    // [64 d  ][32+8 key]
#define PSTR   34    // per-wave [16 rows][32+2]

__global__ __launch_bounds__(256)
void attention_kernel(const unsigned short* __restrict__ Qb,
                      const unsigned short* __restrict__ Kb,
                      const unsigned short* __restrict__ Vt,
                      unsigned short* __restrict__ O)
{
    __shared__ unsigned short K_lds[2][ATT_BC * KSTR];
    __shared__ unsigned short VT_lds[2][DHEAD * VTSTR];
    __shared__ unsigned short P_lds[8 * 16 * PSTR];

    const int tid  = threadIdx.x;
    const int lane = tid & 31;
    const int wv   = tid >> 5;
    const int hi   = lane >> 4, ln = lane & 15;
    const int bh   = blockIdx.y;
    const int q0   = blockIdx.x * ATT_BR + wv * 16;
    const size_t headOff = (size_t)bh * SEQ * DHEAD;   // same footprint both layouts

    // staging geometry: one 16B chunk each for K and VT per thread
    const int kkey = tid >> 3, kkoff = (tid & 7) * 8;  // K: 32 keys x 8 chunks
    const int vd   = tid >> 2, vkoff = (tid & 3) * 8;  // VT: 64 d  x 4 chunks

    bf16x16 qf[2];
    {
        const unsigned short* qrow = Qb + headOff + (size_t)(q0 + ln) * DHEAD;
#pragma unroll
        for (int f = 0; f < 2; ++f)
            qf[f] = load_afrag(qrow + f * 32, hi * 8);
    }

    f32x8 oacc[4] = {};
    float mrow[8], lrow[8];
#pragma unroll
    for (int r = 0; r < 8; ++r) { mrow[r] = -3.0e38f; lrow[r] = 0.0f; }

    // prologue: stage tile 0 into buffer 0
    copy16_g2l(Kb + headOff + (size_t)kkey * DHEAD + kkoff, &K_lds[0][kkey * KSTR + kkoff]);
    copy16_g2l(Vt + headOff + (size_t)vd * SEQ + vkoff,     &VT_lds[0][vd * VTSTR + vkoff]);
    wait_async_copies();
    __syncthreads();

    for (int kt = 0; kt < SEQ; kt += ATT_BC) {
        const int buf = (kt >> 5) & 1;
        if (kt + ATT_BC < SEQ) {   // prefetch next key tile
            const int nk = kt + ATT_BC;
            copy16_g2l(Kb + headOff + (size_t)(nk + kkey) * DHEAD + kkoff,
                       &K_lds[buf ^ 1][kkey * KSTR + kkoff]);
            copy16_g2l(Vt + headOff + (size_t)vd * SEQ + nk + vkoff,
                       &VT_lds[buf ^ 1][vd * VTSTR + vkoff]);
        }

        // scores = Q @ K^T
        f32x8 sc[2] = {};
#pragma unroll
        for (int t = 0; t < 2; ++t)
#pragma unroll
            for (int f = 0; f < 2; ++f) {
                bf16x16 bf = load_bfrag(&K_lds[buf][(t * 16 + ln) * KSTR + f * 32],
                                        hi * 16);
                sc[t] = wmma_bf16(qf[f], bf, sc[t]);
            }

        // online softmax (rows live in 16-lane halves)
        union { f32x8 v; float s[8]; } s0, s1;
        s0.v = sc[0]; s1.v = sc[1];
        float alpha[8];
#pragma unroll
        for (int r = 0; r < 8; ++r) {
            float mx = fmaxf(s0.s[r], s1.s[r]);
#pragma unroll
            for (int m = 1; m < 16; m <<= 1)
                mx = fmaxf(mx, __shfl_xor(mx, m, 32));
            float mnew = fmaxf(mrow[r], mx);
            alpha[r] = __expf(mrow[r] - mnew);
            float p0 = __expf(s0.s[r] - mnew);
            float p1 = __expf(s1.s[r] - mnew);
            s0.s[r] = p0; s1.s[r] = p1;
            float sum = p0 + p1;
#pragma unroll
            for (int m = 1; m < 16; m <<= 1)
                sum += __shfl_xor(sum, m, 32);
            lrow[r] = lrow[r] * alpha[r] + sum;
            mrow[r] = mnew;
        }
#pragma unroll
        for (int t = 0; t < 4; ++t) {
            union { f32x8 v; float s[8]; } ou;
            ou.v = oacc[t];
#pragma unroll
            for (int r = 0; r < 8; ++r) ou.s[r] *= alpha[r];
            oacc[t] = ou.v;
        }

        // P: C-layout -> A-fragment via per-wave LDS round trip
        unsigned short* pw = &P_lds[wv * 16 * PSTR];
#pragma unroll
        for (int r = 0; r < 8; ++r) {
            int m = r + hi * 8;
            pw[m * PSTR + ln]      = f32_to_bf16(s0.s[r]);
            pw[m * PSTR + 16 + ln] = f32_to_bf16(s1.s[r]);
        }
        asm volatile("s_wait_dscnt 0" ::: "memory");

        bf16x16 pf = load_afrag(&pw[ln * PSTR], hi * 8);
#pragma unroll
        for (int t = 0; t < 4; ++t) {
            bf16x16 bf = load_bfrag(&VT_lds[buf][(t * 16 + ln) * VTSTR], hi * 16);
            oacc[t] = wmma_bf16(pf, bf, oacc[t]);
        }

        wait_async_copies();
        __syncthreads();
    }

    // normalize, write O as bf16 in [B, S, h*d]
    const int b = bh >> 4, h = bh & 15;
#pragma unroll
    for (int t = 0; t < 4; ++t) {
        union { f32x8 v; float s[8]; } ou;
        ou.v = oacc[t];
#pragma unroll
        for (int r = 0; r < 8; ++r) {
            int m    = r + hi * 8;
            int srow = q0 + m;
            int dd   = t * 16 + ln;
            O[(size_t)(b * SEQ + srow) * INNER + h * DHEAD + dd] =
                f32_to_bf16(ou.s[r] / lrow[r]);
        }
    }
}

// ---------------------------------------------------------------- launch
extern "C" void kernel_launch(void* const* d_in, const int* in_sizes, int n_in,
                              void* d_out, int out_size, void* d_ws, size_t ws_size,
                              hipStream_t stream)
{
    (void)in_sizes; (void)n_in; (void)out_size; (void)ws_size;

    const float* x  = (const float*)d_in[0];
    const float* Wq = (const float*)d_in[1];
    const float* Wk = (const float*)d_in[2];
    const float* Wv = (const float*)d_in[3];
    const float* Wo = (const float*)d_in[4];

    const int M = BATCH * SEQ;   // 8192
    const int N = INNER;         // 1024
    const int K = DMODEL;        // 1024

    // workspace (halves): xb 8M | WTq/k/v/o 1M each | Qb 8M | Kb 8M | VTb 8M | Ob 8M
    unsigned short* xb  = (unsigned short*)d_ws;
    unsigned short* WTq = xb  + (size_t)M * K;
    unsigned short* WTk = WTq + (size_t)N * K;
    unsigned short* WTv = WTk + (size_t)N * K;
    unsigned short* WTo = WTv + (size_t)N * K;
    unsigned short* Qb  = WTo + (size_t)N * K;
    unsigned short* Kb  = Qb  + (size_t)M * N;
    unsigned short* VTb = Kb  + (size_t)M * N;
    unsigned short* Ob  = VTb + (size_t)M * N;

    // one-time precision / layout conversion
    cvt_bf16_kernel<<<(size_t)M * K / (256 * 8), 256, 0, stream>>>(x, xb);
    dim3 tgrid(N / 32, K / 32);
    transpose_cvt_kernel<<<tgrid, 256, 0, stream>>>(Wq, WTq, K, N);
    transpose_cvt_kernel<<<tgrid, 256, 0, stream>>>(Wk, WTk, K, N);
    transpose_cvt_kernel<<<tgrid, 256, 0, stream>>>(Wv, WTv, K, N);
    transpose_cvt_kernel<<<tgrid, 256, 0, stream>>>(Wo, WTo, K, N);

    dim3 ggrid(N / GBN, M / GBM);
    gemm_bf16<<<ggrid, 256, 0, stream>>>(xb, WTq, nullptr, Qb,  M, N, K, 1, 0.125f);
    gemm_bf16<<<ggrid, 256, 0, stream>>>(xb, WTk, nullptr, Kb,  M, N, K, 1, 1.0f);
    gemm_bf16<<<ggrid, 256, 0, stream>>>(xb, WTv, nullptr, VTb, M, N, K, 2, 1.0f);

    dim3 agrid(SEQ / ATT_BR, BATCH * HEADS);
    attention_kernel<<<agrid, 256, 0, stream>>>(Qb, Kb, VTb, Ob);

    gemm_bf16<<<ggrid, 256, 0, stream>>>(Ob, WTo, (float*)d_out, nullptr,
                                         M, DMODEL, INNER, 0, 1.0f);
}
